// MoE_57380763075167
// MI455X (gfx1250) — compile-verified, matrix operand order checked by reading, not run on hardware
//
#include <hip/hip_runtime.h>
#include <hip/hip_bf16.h>

// Problem constants (match reference)
#define N_TOK 8192
#define DDIM  1024
#define HDIM  4096
#define NEXP  8
#define TOPK  2
#define TMROW 32      // token tile (M): two 16-row WMMA blocks
#define HCHUNK 128    // H chunk width per iteration

typedef __attribute__((ext_vector_type(16))) __bf16        v16bf;
typedef __attribute__((ext_vector_type(8)))  float         v8f;
typedef __attribute__((ext_vector_type(4)))  unsigned int  u32x4;

union FragBF { u32x4 q[2]; unsigned int u[8]; v16bf v; };

__device__ __forceinline__ unsigned short f2bf(float f) {
    union { float f; unsigned int u; } c; c.f = f;
    unsigned int u = c.u;
    unsigned int r = (u + 0x7FFFu + ((u >> 16) & 1u)) >> 16;
    return (unsigned short)r;
}

__device__ __forceinline__ unsigned int pack2(float a, float b) {
    return (unsigned int)f2bf(a) | ((unsigned int)f2bf(b) << 16);
}

__device__ __forceinline__ v8f wmma_bf16(v16bf a, v16bf b, v8f c) {
    return __builtin_amdgcn_wmma_f32_16x16x32_bf16(
        false, a, false, b, (short)0, c, false, false);
}

// ---------------------------------------------------------------------------
// Kernel 0: zero output + expert counters
// ---------------------------------------------------------------------------
__global__ __launch_bounds__(256) void moe_zero_kernel(float* out, int* counts, long total) {
    long i = (long)blockIdx.x * blockDim.x + threadIdx.x;
    if (i < total) out[i] = 0.0f;
    if (i < NEXP)  counts[i] = 0;
}

// ---------------------------------------------------------------------------
// Kernel 1: router (softmax + top-2) + x -> bf16 conversion. One wave32/token.
// ---------------------------------------------------------------------------
__global__ __launch_bounds__(256) void moe_router_kernel(
    const float* __restrict__ x, const float* __restrict__ wr,
    const float* __restrict__ br, unsigned short* __restrict__ xbf,
    int* __restrict__ counts, int* __restrict__ tok_list,
    float* __restrict__ p_list) {

    int n    = blockIdx.x * 8 + (threadIdx.x >> 5);
    int lane = threadIdx.x & 31;
    if (n >= N_TOK) return;

    const float* xrow = x + (long)n * DDIM;
    float acc[NEXP];
#pragma unroll
    for (int e = 0; e < NEXP; ++e) acc[e] = 0.0f;

    for (int j = lane; j < DDIM; j += 32) {
        float xv = xrow[j];
        xbf[(long)n * DDIM + j] = f2bf(xv);
        const float* wrow = wr + (long)j * NEXP;
#pragma unroll
        for (int e = 0; e < NEXP; ++e) acc[e] += xv * wrow[e];
    }
#pragma unroll
    for (int e = 0; e < NEXP; ++e) {
#pragma unroll
        for (int off = 16; off > 0; off >>= 1)
            acc[e] += __shfl_xor(acc[e], off, 32);
    }

    if (lane == 0) {
        float l[NEXP];
        float mx = -3.4e38f;
#pragma unroll
        for (int e = 0; e < NEXP; ++e) { l[e] = acc[e] + br[e]; mx = fmaxf(mx, l[e]); }
        float s = 0.0f;
#pragma unroll
        for (int e = 0; e < NEXP; ++e) { l[e] = __expf(l[e] - mx); s += l[e]; }
        float inv = 1.0f / s;
        int i0 = 0;
#pragma unroll
        for (int e = 1; e < NEXP; ++e) if (l[e] > l[i0]) i0 = e;
        int i1 = (i0 == 0) ? 1 : 0;
#pragma unroll
        for (int e = 0; e < NEXP; ++e) if (e != i0 && l[e] > l[i1]) i1 = e;

        int sel[TOPK] = { i0, i1 };
#pragma unroll
        for (int k = 0; k < TOPK; ++k) {
            int ee = sel[k];
            float p = l[ee] * inv;
            int slot = atomicAdd(&counts[ee], 1);
            tok_list[(long)ee * N_TOK + slot] = n;
            p_list[(long)ee * N_TOK + slot]  = p;
        }
    }
}

// ---------------------------------------------------------------------------
// Weight pre-swizzle: fp32 row-major -> bf16 WMMA B-fragment layout.
// One 32x16 (KxN) fragment per wave; lane L's 8 dwords stored at frag+L*32,
// so the GEMM loop fetches a B fragment with two coalesced global_load_b128.
// ---------------------------------------------------------------------------
__global__ __launch_bounds__(256) void swizzle_w1_kernel(
    const float* __restrict__ w1, unsigned short* __restrict__ w1s) {
    // frag index f = ((e*256 + colTile)*32 + kTile), colTile over H/16, kTile over D/32
    int f    = blockIdx.x * 8 + (threadIdx.x >> 5);
    int lane = threadIdx.x & 31;
    int nn = lane & 15, hl = lane >> 4;
    int ks = f & 31, ct = (f >> 5) & 255, e = f >> 13;
    const float* base = w1 + (long)e * DDIM * HDIM + (long)(ks * 32) * HDIM + ct * 16 + nn;
    FragBF B;
#pragma unroll
    for (int j = 0; j < 8; ++j) {
        int koff = (j < 4) ? (hl * 8 + 2 * j) : (16 + hl * 8 + 2 * (j - 4));
        const float* p = base + (long)koff * HDIM;
        B.u[j] = pack2(p[0], p[HDIM]);
    }
    u32x4* dst = (u32x4*)((char*)w1s + (long)f * 1024 + lane * 32);
    dst[0] = B.q[0]; dst[1] = B.q[1];
}

__global__ __launch_bounds__(256) void swizzle_w2_kernel(
    const float* __restrict__ w2, unsigned short* __restrict__ w2s) {
    // frag index f = ((e*64 + colTile)*128 + kTile), colTile over D/16, kTile over H/32
    int f    = blockIdx.x * 8 + (threadIdx.x >> 5);
    int lane = threadIdx.x & 31;
    int nn = lane & 15, hl = lane >> 4;
    int ks = f & 127, ct = (f >> 7) & 63, e = f >> 13;
    const float* base = w2 + (long)e * HDIM * DDIM + (long)(ks * 32) * DDIM + ct * 16 + nn;
    FragBF B;
#pragma unroll
    for (int j = 0; j < 8; ++j) {
        int koff = (j < 4) ? (hl * 8 + 2 * j) : (16 + hl * 8 + 2 * (j - 4));
        const float* p = base + (long)koff * DDIM;
        B.u[j] = pack2(p[0], p[DDIM]);
    }
    u32x4* dst = (u32x4*)((char*)w2s + (long)f * 1024 + lane * 32);
    dst[0] = B.q[0]; dst[1] = B.q[1];
}

// ---------------------------------------------------------------------------
// Kernel 2: grouped expert FFN. grid=(256 token-tiles, 8 experts), 8 waves.
// 32 tokens/tile (two 16-row WMMA blocks share each B fragment).
// A fragments: 2x ds_load_b128 from swizzled LDS. B: 2x global_load_b128.
// ---------------------------------------------------------------------------
__global__ __launch_bounds__(256) void moe_ffn_kernel(
    const unsigned short* __restrict__ xbf,
    const unsigned short* __restrict__ w1s, const float* __restrict__ b1,
    const unsigned short* __restrict__ w2s, const float* __restrict__ b2,
    const int* __restrict__ counts, const int* __restrict__ tok_list,
    const float* __restrict__ p_list, float* __restrict__ out) {

    __shared__ unsigned char xs_sw[32 * 2 * 32 * 32]; // [kTile][rb][lane][32B] = 64KB
    __shared__ unsigned char hs_sw[4 * 2 * 32 * 32];  // [kTile][rb][lane][32B] = 8KB
    __shared__ int   toks[TMROW];
    __shared__ float probs[TMROW];

    const int e    = blockIdx.y;
    const int tile = blockIdx.x;
    const int cnt  = counts[e];
    if (tile * TMROW >= cnt) return;

    const int tid  = threadIdx.x;
    const int wave = tid >> 5;
    const int lane = tid & 31;
    const int hl   = lane >> 4;
    const int nn   = lane & 15;

    if (tid < TMROW) {
        int idx = tile * TMROW + tid;
        if (idx < cnt) {
            toks[tid]  = tok_list[(long)e * N_TOK + idx];
            probs[tid] = p_list[(long)e * N_TOK + idx];
        } else {
            toks[tid]  = tok_list[(long)e * N_TOK + cnt - 1]; // pad rows, prob=0
            probs[tid] = 0.0f;
        }
    }
    __syncthreads();

    // Stage x tile into LDS in A-fragment layout (pure 16B copies):
    // src chunk for (kTile, laneHalf lh, q): bytes kTile*64 + q*32 + lh*16 of the row.
    for (int i = tid; i < 4096; i += 256) {
        int q  = i & 1;
        int ln = (i >> 1) & 31;
        int rb = (i >> 6) & 1;
        int ks = i >> 7;
        int m  = ln & 15, lh = ln >> 4;
        const char* src = (const char*)(xbf + (long)toks[rb * 16 + m] * DDIM)
                          + ks * 64 + q * 32 + lh * 16;
        char* dst = (char*)xs_sw + (((ks * 2 + rb) * 32 + ln) * 32) + q * 16;
        *(u32x4*)dst = *(const u32x4*)src;
    }
    __syncthreads();

    v8f acc0[8], acc1[8];
#pragma unroll
    for (int t = 0; t < 8; ++t)
#pragma unroll
        for (int r = 0; r < 8; ++r) { acc0[t][r] = 0.0f; acc1[t][r] = 0.0f; }

    // Precompute this lane's h-store swizzle (its h column within the chunk is fixed)
    const int kcol   = wave * 16 + nn;        // 0..127 within chunk
    const int ks2    = kcol >> 5;
    const int klocal = kcol & 31;
    const int r2     = (klocal < 16) ? klocal : (klocal - 16);
    const int hlp    = r2 >> 3;
    const int jj     = ((r2 & 7) >> 1) + ((klocal < 16) ? 0 : 4);
    const int bb     = klocal & 1;

    for (int hc = 0; hc < HDIM / HCHUNK; ++hc) {
        // ---- GEMM1: wave computes 16 h-columns x 32 tokens ----
        v8f h0, h1;
#pragma unroll
        for (int r = 0; r < 8; ++r) { h0[r] = 0.0f; h1[r] = 0.0f; }

        const char* w1base = (const char*)w1s
            + ((long)(e * 256 + hc * 8 + wave) * 32) * 1024 + lane * 32;
        for (int ks = 0; ks < DDIM / 32; ++ks) {
            FragBF A0, A1, B;
            const u32x4* bp = (const u32x4*)(w1base + (long)ks * 1024);
            B.q[0] = bp[0]; B.q[1] = bp[1];
            const u32x4* a0 = (const u32x4*)(xs_sw + ((ks * 2 + 0) * 32 + lane) * 32);
            const u32x4* a1 = (const u32x4*)(xs_sw + ((ks * 2 + 1) * 32 + lane) * 32);
            A0.q[0] = a0[0]; A0.q[1] = a0[1];
            A1.q[0] = a1[0]; A1.q[1] = a1[1];
            h0 = wmma_bf16(A0.v, B.v, h0);
            h1 = wmma_bf16(A1.v, B.v, h1);
        }
        // bias + exact GELU; store bf16 h into LDS in A-fragment layout
        const float bias1 = b1[(long)e * HDIM + hc * HCHUNK + kcol];
#pragma unroll
        for (int r = 0; r < 8; ++r) {
            const int m = r + 8 * hl;
            float v0 = h0[r] + bias1;
            float v1 = h1[r] + bias1;
            v0 = 0.5f * v0 * (1.0f + erff(v0 * 0.70710678118654752f));
            v1 = 0.5f * v1 * (1.0f + erff(v1 * 0.70710678118654752f));
            const int base = (hlp * 16 + m) * 32 + jj * 4 + bb * 2;
            *(unsigned short*)(hs_sw + ((ks2 * 2 + 0) * 32) * 32 + base) = f2bf(v0);
            *(unsigned short*)(hs_sw + ((ks2 * 2 + 1) * 32) * 32 + base) = f2bf(v1);
        }
        __syncthreads();

        // ---- GEMM2: wave accumulates its 128 output columns x 32 tokens ----
#pragma unroll
        for (int t = 0; t < 8; ++t) {
            const char* w2base = (const char*)w2s
                + ((long)((e * 64 + wave * 8 + t) * 128 + hc * 4)) * 1024 + lane * 32;
#pragma unroll
            for (int ks = 0; ks < 4; ++ks) {
                FragBF A0, A1, B;
                const u32x4* bp = (const u32x4*)(w2base + (long)ks * 1024);
                B.q[0] = bp[0]; B.q[1] = bp[1];
                const u32x4* a0 = (const u32x4*)(hs_sw + ((ks * 2 + 0) * 32 + lane) * 32);
                const u32x4* a1 = (const u32x4*)(hs_sw + ((ks * 2 + 1) * 32 + lane) * 32);
                A0.q[0] = a0[0]; A0.q[1] = a0[1];
                A1.q[0] = a1[0]; A1.q[1] = a1[1];
                acc0[t] = wmma_bf16(A0.v, B.v, acc0[t]);
                acc1[t] = wmma_bf16(A1.v, B.v, acc1[t]);
            }
        }
        __syncthreads();
    }

    // ---- Scatter: out[token] += prob * (acc + b2) ----
#pragma unroll
    for (int t = 0; t < 8; ++t) {
        const int col = wave * 128 + t * 16 + nn;
        const float bias2 = b2[(long)e * DDIM + col];
#pragma unroll
        for (int r = 0; r < 8; ++r) {
            const int m = r + 8 * hl;
            float v0 = (acc0[t][r] + bias2) * probs[m];
            float v1 = (acc1[t][r] + bias2) * probs[16 + m];
            atomicAdd(&out[(long)toks[m]      * DDIM + col], v0);
            atomicAdd(&out[(long)toks[16 + m] * DDIM + col], v1);
        }
    }
}

// ---------------------------------------------------------------------------
// Host launcher
// ---------------------------------------------------------------------------
extern "C" void kernel_launch(void* const* d_in, const int* in_sizes, int n_in,
                              void* d_out, int out_size, void* d_ws, size_t ws_size,
                              hipStream_t stream) {
    const float* x  = (const float*)d_in[0];  // [N, D]
    const float* wr = (const float*)d_in[1];  // [D, E]
    const float* br = (const float*)d_in[2];  // [E]
    const float* w1 = (const float*)d_in[3];  // [E, D, H]
    const float* b1 = (const float*)d_in[4];  // [E, H]
    const float* w2 = (const float*)d_in[5];  // [E, H, D]
    const float* b2 = (const float*)d_in[6];  // [E, D]
    float* out = (float*)d_out;               // [N, D]

    // Workspace: counts | token lists | prob lists | bf16 x | swizzled w1 | swizzled w2
    int*   counts = (int*)d_ws;
    int*   tok    = counts + 16;
    float* plist  = (float*)(tok + (long)NEXP * N_TOK);
    unsigned short* xbf = (unsigned short*)(plist + (long)NEXP * N_TOK);   // N*D bf16
    unsigned short* w1s = xbf + (long)N_TOK * DDIM;                        // 64MB
    unsigned short* w2s = w1s + (long)NEXP * DDIM * HDIM;                  // 64MB

    const long total = (long)N_TOK * DDIM;
    moe_zero_kernel<<<(int)((total + 255) / 256), 256, 0, stream>>>(out, counts, total);

    moe_router_kernel<<<N_TOK / 8, 256, 0, stream>>>(x, wr, br, xbf, counts, tok, plist);

    // 65536 fragments each; 8 waves per block
    swizzle_w1_kernel<<<8192, 256, 0, stream>>>(w1, w1s);
    swizzle_w2_kernel<<<8192, 256, 0, stream>>>(w2, w2s);

    dim3 grid(N_TOK / TMROW, NEXP);
    moe_ffn_kernel<<<grid, 256, 0, stream>>>(xbf, w1s, b1, w2s, b2,
                                             counts, tok, plist, out);
}
